// OriginalModel_gcn_26482768347991
// MI455X (gfx1250) — matrix operand order repaired and checked by reading, not compiled, not run on hardware
//
#include <hip/hip_runtime.h>
#include <hip/hip_bf16.h>
#include <stdint.h>

typedef __attribute__((ext_vector_type(16))) _Float16 v16h;
typedef __attribute__((ext_vector_type(8)))  float    v8f;

union FragH {
  v16h v;
  uint32_t u[8];
  uint4 q[2];
};
union FragF {
  v8f v;
  float f[8];
};

__device__ inline v8f wmma16(const v16h& a, const v16h& b, const v8f& c) {
  // D = A(16x32 f16) * B(32x16 f16) + C(16x16 f32)
  return __builtin_amdgcn_wmma_f32_16x16x32_f16(false, a, false, b, (short)0, c,
                                                false, false);
}

// A fragment: per-lane 16 f16 = two contiguous 8-halves at +khA and +khA+16.
// 4-byte-aligned path (conv1: row stride 6 halves = 12B).
__device__ inline void ldA_u32(const _Float16* p, FragH& f) {
  const uint32_t* q0 = reinterpret_cast<const uint32_t*>(p);
  const uint32_t* q1 = reinterpret_cast<const uint32_t*>(p + 16);
  f.u[0] = q0[0]; f.u[1] = q0[1]; f.u[2] = q0[2]; f.u[3] = q0[3];
  f.u[4] = q1[0]; f.u[5] = q1[1]; f.u[6] = q1[2]; f.u[7] = q1[3];
}
// 16-byte-aligned path (conv2 / GEMMs): two global_load_b128.
__device__ inline void ldA_u4(const _Float16* p, FragH& f) {
  f.q[0] = *reinterpret_cast<const uint4*>(p);
  f.q[1] = *reinterpret_cast<const uint4*>(p + 16);
}
// B fragment from pre-transposed weights [N,K]: two contiguous 8-halves at +khB and +khB+16.
__device__ inline void ldB_u4(const _Float16* p, FragH& f) {
  const uint4* q = reinterpret_cast<const uint4*>(p);
  f.q[0] = q[0]; f.q[1] = q[2];
}

__device__ inline v8f zero8() {
  v8f z = {0.f, 0.f, 0.f, 0.f, 0.f, 0.f, 0.f, 0.f};
  return z;
}

// ---------------------------------------------------------------------------
// GCN normalization: norm[i,j] = dinv[i] * Ahat[j,i] * dinv[j]
// ---------------------------------------------------------------------------
__global__ __launch_bounds__(64)
void norm_kernel(const float* __restrict__ A, float* __restrict__ nrm) {
  __shared__ float dinv[39];
  const int t = threadIdx.x;
  if (t < 39) {
    float s = 0.f;
    for (int i = 0; i < 39; ++i) s += A[i * 39 + t];   // column sum (deg)
    if (A[t * 39 + t] == 0.f) s += 1.f;                // remaining self loop
    dinv[t] = (s > 0.f) ? (1.0f / sqrtf(s)) : 0.f;
  }
  __syncthreads();
  for (int i = t; i < 39 * 39; i += 64) {
    const int r = i / 39, c = i % 39;
    float a = A[c * 39 + r];                           // Ahat[j=c, i=r]
    if (r == c && a == 0.f) a = 1.f;
    nrm[i] = dinv[r] * a * dinv[c];
  }
}

// ---------------------------------------------------------------------------
// Weight packing (transposed [Cout, K] f16, K ordered to match A windows)
// ---------------------------------------------------------------------------
__global__ __launch_bounds__(256)
void pack_w1(const float* __restrict__ W, _Float16* __restrict__ Bt) {
  const int idx = blockIdx.x * 256 + threadIdx.x;
  if (idx >= 32 * 384) return;
  const int oc = idx / 384, kk = idx % 384;
  const int k = kk / 3, ic = kk % 3;                   // K-linear = k*3+ic
  const float v = (k < 125) ? W[(oc * 3 + ic) * 125 + k] : 0.f;
  Bt[idx] = (_Float16)v;
}

__global__ __launch_bounds__(256)
void pack_w2(const float* __restrict__ W, _Float16* __restrict__ Bt) {
  const int idx = blockIdx.x * 256 + threadIdx.x;
  if (idx >= 64 * 4000) return;
  const int oc = idx / 4000, kk = idx % 4000;
  const int k = kk / 32, ic = kk % 32;                 // K-linear = k*32+ic
  Bt[idx] = (_Float16)W[(oc * 32 + ic) * 125 + k];
}

__global__ __launch_bounds__(256)
void pack_g1(const float* __restrict__ W, _Float16* __restrict__ Bt) {
  const int idx = blockIdx.x * 256 + threadIdx.x;
  if (idx >= 64 * 10048) return;
  const int n = idx / 10048, k = idx % 10048;
  Bt[idx] = (_Float16)W[(size_t)k * 64 + n];
}

__global__ __launch_bounds__(256)
void pack_fc(const float* __restrict__ W, _Float16* __restrict__ Bt) {
  const int idx = blockIdx.x * 256 + threadIdx.x;
  if (idx >= 128 * 2496) return;
  const int n = idx / 2496, k = idx % 2496;
  Bt[idx] = (_Float16)W[(size_t)k * 128 + n];
}

__global__ __launch_bounds__(256)
void cvt_x(const float* __restrict__ x, _Float16* __restrict__ xh, int n, int npad) {
  const int idx = blockIdx.x * 256 + threadIdx.x;
  if (idx >= npad) return;
  xh[idx] = (idx < n) ? (_Float16)x[idx] : (_Float16)0.f;
}

// ---------------------------------------------------------------------------
// Conv1: implicit GEMM. Rows = (b,n) signals [2496]. M=438 out positions,
// N=32, K=375 (padded 384). A row m: contiguous 375 f16 at offset 6*m.
// Wave tile: 32(M) x 32(N)  (MT=2, NT=2).
// Out: y1[row, ow, oc] position-major f16 (post-ReLU+bias).
// ---------------------------------------------------------------------------
__global__ __launch_bounds__(32)
void conv1_wmma(const _Float16* __restrict__ X, const _Float16* __restrict__ Bt,
                const float* __restrict__ bias, _Float16* __restrict__ Y) {
  const int row = blockIdx.x;            // b*39+n
  const int mbase = blockIdx.y * 32;     // output position tile (MT=2)
  const int lane = threadIdx.x;
  const int ml = lane & 15;
  const int khA = (lane >> 4) * 8;
  const int khB = (lane >> 4) * 8;
  const _Float16* Arow0 = X + (size_t)row * 3000 + (size_t)(mbase + ml) * 6;
  const _Float16* Arow1 = Arow0 + 16 * 6;
  const _Float16* B0 = Bt + (size_t)ml * 384;
  const _Float16* B1 = Bt + (size_t)(ml + 16) * 384;
  FragF c[2][2];
#pragma unroll
  for (int mt = 0; mt < 2; ++mt)
    for (int t = 0; t < 2; ++t) c[mt][t].v = zero8();
  for (int kit = 0; kit < 12; ++kit) {
    const int ko = kit * 32;
    FragH a0, a1;
    ldA_u32(Arow0 + ko + khA, a0);
    ldA_u32(Arow1 + ko + khA, a1);
    FragH b0, b1;
    ldB_u4(B0 + ko + khB, b0);
    ldB_u4(B1 + ko + khB, b1);
    c[0][0].v = wmma16(a0.v, b0.v, c[0][0].v);
    c[0][1].v = wmma16(a0.v, b1.v, c[0][1].v);
    c[1][0].v = wmma16(a1.v, b0.v, c[1][0].v);
    c[1][1].v = wmma16(a1.v, b1.v, c[1][1].v);
  }
  const int hi = (lane >> 4) * 8;
  const float bo0 = bias[ml], bo1 = bias[ml + 16];
  _Float16* Yrow = Y + (size_t)row * (438 * 32);
#pragma unroll
  for (int mt = 0; mt < 2; ++mt) {
#pragma unroll
    for (int r = 0; r < 8; ++r) {
      const int ow = mbase + mt * 16 + hi + r;
      if (ow < 438) {
        Yrow[(size_t)ow * 32 + ml]      = (_Float16)fmaxf(c[mt][0].f[r] + bo0, 0.f);
        Yrow[(size_t)ow * 32 + ml + 16] = (_Float16)fmaxf(c[mt][1].f[r] + bo1, 0.f);
      }
    }
  }
}

// ---------------------------------------------------------------------------
// Conv2: M=157 positions, N=64, K=4000 (=125*32, exact). A row m: contiguous
// 4000 f16 at offset 64*m of y1 row. Wave tile: 32(M) x 64(N) (MT=2, NT=4).
// Output written NCHW f16 (post-ReLU), which IS the reshape(B,39,10048)
// GEMM-A for Wg1.
// ---------------------------------------------------------------------------
__global__ __launch_bounds__(32)
void conv2_wmma(const _Float16* __restrict__ Y1, const _Float16* __restrict__ Bt,
                const float* __restrict__ bias, _Float16* __restrict__ Y2) {
  const int row = blockIdx.x;            // b*39+n
  const int mbase = blockIdx.y * 32;     // 0,32,64,96,128 (covers 160)
  const int lane = threadIdx.x;
  const int ml = lane & 15;
  const int khA = (lane >> 4) * 8;
  const int khB = (lane >> 4) * 8;
  const _Float16* Arow0 = Y1 + (size_t)row * 14016 + (size_t)(mbase + ml) * 64;
  const _Float16* Arow1 = Arow0 + 16 * 64;
  const _Float16* Bp[4];
  FragF c[2][4];
#pragma unroll
  for (int t = 0; t < 4; ++t) {
    Bp[t] = Bt + (size_t)(t * 16 + ml) * 4000;
    c[0][t].v = zero8();
    c[1][t].v = zero8();
  }
  for (int kit = 0; kit < 125; ++kit) {
    const int ko = kit * 32;
    FragH a0, a1;
    ldA_u4(Arow0 + ko + khA, a0);
    ldA_u4(Arow1 + ko + khA, a1);
#pragma unroll
    for (int t = 0; t < 4; ++t) {
      FragH b; ldB_u4(Bp[t] + ko + khB, b);
      c[0][t].v = wmma16(a0.v, b.v, c[0][t].v);
      c[1][t].v = wmma16(a1.v, b.v, c[1][t].v);
    }
  }
  const int b = row / 39, nn = row % 39;
  const int hi = (lane >> 4) * 8;
#pragma unroll
  for (int t = 0; t < 4; ++t) {
    const int oc = t * 16 + ml;
    const float bo = bias[oc];
    _Float16* dst = Y2 + ((size_t)(b * 64 + oc) * 39 + nn) * 157;
#pragma unroll
    for (int mt = 0; mt < 2; ++mt) {
#pragma unroll
      for (int r = 0; r < 8; ++r) {
        const int ow = mbase + mt * 16 + hi + r;
        if (ow < 157) dst[ow] = (_Float16)fmaxf(c[mt][t].f[r] + bo, 0.f);
      }
    }
  }
}

// ---------------------------------------------------------------------------
// Generic f16 WMMA GEMM: C[M,N] = A[M,K] * Bt[N,K]^T (+bias, relu). One wave
// per (MT*16) x (NT*16) tile. K multiple of 32; M multiple of MT*16.
// ---------------------------------------------------------------------------
template <int MT, int NT>
__global__ __launch_bounds__(32)
void gemm_wmma(const _Float16* __restrict__ A, int lda,
               const _Float16* __restrict__ Bt, int ldb,
               float* __restrict__ C, int ldc,
               const float* __restrict__ bias, int relu, int kIters) {
  const int mbase = blockIdx.x * (MT * 16);
  const int nbase = blockIdx.y * (NT * 16);
  const int lane = threadIdx.x;
  const int ml = lane & 15;
  const int khA = (lane >> 4) * 8;
  const int khB = (lane >> 4) * 8;
  const _Float16* Arow[MT];
  const _Float16* Bp[NT];
  FragF c[MT][NT];
#pragma unroll
  for (int mt = 0; mt < MT; ++mt)
    Arow[mt] = A + (size_t)(mbase + mt * 16 + ml) * lda;
#pragma unroll
  for (int t = 0; t < NT; ++t) {
    Bp[t] = Bt + (size_t)(nbase + t * 16 + ml) * ldb;
#pragma unroll
    for (int mt = 0; mt < MT; ++mt) c[mt][t].v = zero8();
  }
  for (int kit = 0; kit < kIters; ++kit) {
    const int ko = kit * 32;
    FragH a[MT];
#pragma unroll
    for (int mt = 0; mt < MT; ++mt) ldA_u4(Arow[mt] + ko + khA, a[mt]);
#pragma unroll
    for (int t = 0; t < NT; ++t) {
      FragH b; ldB_u4(Bp[t] + ko + khB, b);
#pragma unroll
      for (int mt = 0; mt < MT; ++mt)
        c[mt][t].v = wmma16(a[mt].v, b.v, c[mt][t].v);
    }
  }
  const int hi = (lane >> 4) * 8;
#pragma unroll
  for (int t = 0; t < NT; ++t) {
    const int n = nbase + t * 16 + ml;
    const float bo = bias ? bias[n] : 0.f;
#pragma unroll
    for (int mt = 0; mt < MT; ++mt) {
#pragma unroll
      for (int r = 0; r < 8; ++r) {
        const int m = mbase + mt * 16 + hi + r;
        float v = c[mt][t].f[r] + bo;
        if (relu) v = fmaxf(v, 0.f);
        C[(size_t)m * ldc + n] = v;
      }
    }
  }
}

// ---------------------------------------------------------------------------
// GCN block (per batch): h = relu(norm@z1 + bg1); hw = h@Wg2;
// x2 = tanh(norm@hw + bg2) -> f16 [b,2496]. All in LDS; tiny K.
// ---------------------------------------------------------------------------
__global__ __launch_bounds__(256)
void gcn_kernel(const float* __restrict__ z1, const float* __restrict__ nrm,
                const float* __restrict__ bg1, const float* __restrict__ Wg2,
                const float* __restrict__ bg2, _Float16* __restrict__ x2h) {
  __shared__ float sN[39 * 39];
  __shared__ float sZ[39 * 64];
  __shared__ float sH[39 * 64];
  __shared__ float sHW[39 * 64];
  __shared__ float sW[64 * 64];
  const int b = blockIdx.x;
  const int t = threadIdx.x;
  for (int i = t; i < 39 * 39; i += 256) sN[i] = nrm[i];
  for (int i = t; i < 2496; i += 256) sZ[i] = z1[(size_t)b * 2496 + i];
  for (int i = t; i < 4096; i += 256) sW[i] = Wg2[i];
  __syncthreads();
  for (int i = t; i < 2496; i += 256) {
    const int node = i >> 6, f = i & 63;
    float acc = bg1[f];
    for (int m = 0; m < 39; ++m) acc += sN[node * 39 + m] * sZ[m * 64 + f];
    sH[i] = fmaxf(acc, 0.f);
  }
  __syncthreads();
  for (int i = t; i < 2496; i += 256) {
    const int node = i >> 6, f = i & 63;
    float acc = 0.f;
    for (int c = 0; c < 64; ++c) acc += sH[node * 64 + c] * sW[c * 64 + f];
    sHW[i] = acc;
  }
  __syncthreads();
  for (int i = t; i < 2496; i += 256) {
    const int node = i >> 6, f = i & 63;
    float acc = bg2[f];
    for (int m = 0; m < 39; ++m) acc += sN[node * 39 + m] * sHW[m * 64 + f];
    x2h[(size_t)b * 2496 + i] = (_Float16)tanhf(acc);
  }
}

// ---------------------------------------------------------------------------
// Five linear heads: out[h][m][j] = xfc[m,:] @ Wh[:,j] + bh[j]
// ---------------------------------------------------------------------------
__global__ __launch_bounds__(256)
void heads_kernel(const float* __restrict__ xfc,
                  const float* W0, const float* B0, const float* W1, const float* B1,
                  const float* W2, const float* B2, const float* W3, const float* B3,
                  const float* W4, const float* B4, float* __restrict__ out) {
  const int idx = blockIdx.x * 256 + threadIdx.x;
  if (idx >= 5 * 64 * 39) return;
  const int h = idx / 2496, rem = idx % 2496;
  const int m = rem / 39, j = rem % 39;
  const float* W; const float* bb;
  switch (h) {
    case 0: W = W0; bb = B0; break;
    case 1: W = W1; bb = B1; break;
    case 2: W = W2; bb = B2; break;
    case 3: W = W3; bb = B3; break;
    default: W = W4; bb = B4; break;
  }
  float acc = bb[j];
  const float* xr = xfc + (size_t)m * 128;
  for (int c = 0; c < 128; ++c) acc += xr[c] * W[c * 39 + j];
  out[idx] = acc;
}

// ---------------------------------------------------------------------------
extern "C" void kernel_launch(void* const* d_in, const int* in_sizes, int n_in,
                              void* d_out, int out_size, void* d_ws, size_t ws_size,
                              hipStream_t stream) {
  (void)in_sizes; (void)n_in; (void)out_size; (void)ws_size;
  const float* wav = (const float*)d_in[0];
  // d_in[1] graph_features: unused by the reference
  const float* A   = (const float*)d_in[2];
  const float* W1  = (const float*)d_in[3];  const float* b1  = (const float*)d_in[4];
  const float* W2  = (const float*)d_in[5];  const float* b2  = (const float*)d_in[6];
  const float* Wg1 = (const float*)d_in[7];  const float* bg1 = (const float*)d_in[8];
  const float* Wg2 = (const float*)d_in[9];  const float* bg2 = (const float*)d_in[10];
  const float* Wfc = (const float*)d_in[11]; const float* bfc = (const float*)d_in[12];
  float* out = (float*)d_out;

  uintptr_t p = (uintptr_t)d_ws;
  auto alloc = [&](size_t bytes) -> void* {
    p = (p + 255) & ~(uintptr_t)255;
    void* r = (void*)p;
    p += bytes;
    return r;
  };
  float*    normM = (float*)alloc(39 * 39 * 4);
  _Float16* B1t   = (_Float16*)alloc((size_t)32 * 384 * 2);
  _Float16* B2t   = (_Float16*)alloc((size_t)64 * 4000 * 2);
  _Float16* G1t   = (_Float16*)alloc((size_t)64 * 10048 * 2);
  _Float16* Ft    = (_Float16*)alloc((size_t)128 * 2496 * 2);
  _Float16* Xh    = (_Float16*)alloc(((size_t)7488000 + 128) * 2);
  _Float16* Y1    = (_Float16*)alloc(((size_t)2496 * 14016 + 256) * 2);
  _Float16* Y2    = (_Float16*)alloc((size_t)2496 * 10048 * 2);
  float*    Z1    = (float*)alloc((size_t)2496 * 64 * 4);
  _Float16* X2h   = (_Float16*)alloc((size_t)64 * 2496 * 2);
  float*    Xfc   = (float*)alloc((size_t)64 * 128 * 4);

  norm_kernel<<<1, 64, 0, stream>>>(A, normM);
  pack_w1<<<(32 * 384 + 255) / 256, 256, 0, stream>>>(W1, B1t);
  pack_w2<<<(64 * 4000 + 255) / 256, 256, 0, stream>>>(W2, B2t);
  pack_g1<<<(64 * 10048 + 255) / 256, 256, 0, stream>>>(Wg1, G1t);
  pack_fc<<<(128 * 2496 + 255) / 256, 256, 0, stream>>>(Wfc, Ft);
  cvt_x<<<(7488128 + 255) / 256, 256, 0, stream>>>(wav, Xh, 7488000, 7488128);

  // 32x32 tiles over 438 positions -> 14 m-blocks
  conv1_wmma<<<dim3(2496, 14), 32, 0, stream>>>(Xh, B1t, b1, Y1);
  // 32x64 tiles over 157 positions -> 5 m-blocks
  conv2_wmma<<<dim3(2496, 5), 32, 0, stream>>>(Y1, B2t, b2, Y2);
  // x @ Wg1 : [2496,10048] x [10048,64]; 32x64 tiles -> 78 m-blocks
  gemm_wmma<2, 4><<<dim3(78, 1), 32, 0, stream>>>(Y2, 10048, G1t, 10048, Z1, 64,
                                                  nullptr, 0, 314);
  gcn_kernel<<<64, 256, 0, stream>>>(Z1, normM, bg1, Wg2, bg2, X2h);
  // x @ Wfc : [64,2496] x [2496,128], +bias, ReLU
  gemm_wmma<1, 4><<<dim3(4, 2), 32, 0, stream>>>(X2h, 2496, Ft, 2496, Xfc, 128,
                                                 bfc, 1, 78);
  heads_kernel<<<(12480 + 255) / 256, 256, 0, stream>>>(
      Xfc, (const float*)d_in[13], (const float*)d_in[14],
      (const float*)d_in[15], (const float*)d_in[16],
      (const float*)d_in[17], (const float*)d_in[18],
      (const float*)d_in[19], (const float*)d_in[20],
      (const float*)d_in[21], (const float*)d_in[22], out);
}
